// CrossAttentionMultiLayerPerceptron_61976378082028
// MI455X (gfx1250) — matrix operand order repaired and checked by reading, not compiled, not run on hardware
//
#include <hip/hip_runtime.h>
#include <hip/hip_fp16.h>

typedef _Float16 half_t;
typedef __attribute__((ext_vector_type(16))) _Float16 v16h;
typedef __attribute__((ext_vector_type(8)))  float    v8f;
typedef __attribute__((ext_vector_type(4))) unsigned int u32x4;
typedef __attribute__((ext_vector_type(8))) int i32x8;
typedef __attribute__((ext_vector_type(4))) int i32x4;

static constexpr int INDIM = 128;   // input dim
static constexpr int FD    = 256;   // feature dim
static constexpr int HID   = 256;   // hidden dim
static constexpr int OUTD  = 128;   // output dim
static constexpr int TXT   = 512;   // text emb dim

#if __has_builtin(__builtin_amdgcn_tensor_load_to_lds) && __has_builtin(__builtin_amdgcn_s_wait_tensorcnt)
#define USE_TDM 1
#else
#define USE_TDM 0
#endif

// K index for 16-bit A-matrix 16x32 fragment (ISA 7.12.2):
// VGPR i holds (lanes0-15) K = {2i,2i+1} for i<4, {16+2(i-4),+1} for i>=4;
// lanes16-31 add +8.
__device__ __forceinline__ int a_kmap(int j, int lanehi) {
  int i = j >> 1;
  int base = (i < 4) ? (2 * i) : (16 + 2 * (i - 4));
  return base + (lanehi ? 8 : 0) + (j & 1);
}

#if USE_TDM
// One TDM 1-D load per wave: nrows*INDIM f32, contiguous in global, written to
// LDS with +1 DWORD pad every 128 DWORDs (pitch 129) for conflict-free gathers.
__device__ __forceinline__ void tdm_stage_tile(const float* gsrc, float* lds_dst,
                                               unsigned valid_elems) {
  unsigned long long ga = (unsigned long long)(const void*)gsrc;
  unsigned lds_base = (unsigned)(size_t)(void*)lds_dst;
  u32x4 g0 = {1u,                                   // count=1 (valid descriptor)
              lds_base,                             // lds_addr (bytes)
              (unsigned)(ga & 0xFFFFFFFFu),         // global_addr[31:0]
              (unsigned)((ga >> 32) & 0x01FFFFFFu) | 0x80000000u}; // [56:32] | type=2
  i32x8 g1;
  g1[0] = (2 << 16) | (1 << 20) | (6 << 22);        // data_size=4B, pad_en, pad every 128DW by 1DW
  g1[1] = (int)((valid_elems & 0xFFFFu) << 16);     // tensor_dim0[15:0]
  g1[2] = (int)((valid_elems >> 16) | (1u << 16)); // tensor_dim0[31:16] | tensor_dim1=1
  g1[3] = (int)((unsigned)(16 * INDIM) << 16);      // tile_dim0 = 2048 (OOB reads -> 0)
  g1[4] = 0;                                        // tile_dim1=0 (1-D), tile_dim2=0
  g1[5] = (int)valid_elems;                         // tensor_dim0_stride[31:0]
  g1[6] = 0; g1[7] = 0;
  i32x4 gz = {0, 0, 0, 0};
#if __clang_major__ >= 23
  i32x8 gz8 = {0, 0, 0, 0, 0, 0, 0, 0};
  __builtin_amdgcn_tensor_load_to_lds(g0, g1, gz, gz, gz8, 0);
#else
  __builtin_amdgcn_tensor_load_to_lds(g0, g1, gz, gz, 0);
#endif
}
#endif

// ---------------- K0: weight fragment swizzle + hsum zero ----------------
// B-fragment layout (K x N, f16, 16x16x32): lanes0-15 -> N=lane, K=ks*32+a_kmap(j,0);
// lanes16-31 -> N=lane-16, K=ks*32+a_kmap(j,1)   (j = 0..15 halves per lane)
__global__ void k0_prep(const float* __restrict__ W0, const float* __restrict__ W2,
                        half_t* __restrict__ w0frag, half_t* __restrict__ w2frag,
                        float* __restrict__ hsum, int hsumN) {
  int idx = blockIdx.x * blockDim.x + threadIdx.x;
  if (idx < 16 * 4 * 32 * 16) {   // W0: nt(16) x ks(4) x lane(32) x j(16)
    int j = idx & 15, lane = (idx >> 4) & 31, ks = (idx >> 9) & 3, nt = idx >> 11;
    int n = nt * 16 + (lane & 15);
    int k = ks * 32 + a_kmap(j, lane >= 16);
    w0frag[idx] = (half_t)W0[k * FD + n];
  }
  if (idx < 8 * 8 * 32 * 16) {    // W2: ct(8) x ks(8) x lane(32) x j(16)
    int j = idx & 15, lane = (idx >> 4) & 31, ks = (idx >> 9) & 7, ct = idx >> 12;
    int n = ct * 16 + (lane & 15);
    int k = ks * 32 + a_kmap(j, lane >= 16);
    w2frag[idx] = (half_t)W2[k * OUTD + n];
  }
  if (idx < hsumN) hsum[idx] = 0.f;
}

// ---------------- K1: per-graph q, wscore = Wk@q, c = q.bk, offsets, seg ---
__global__ void k1_graph(const float* __restrict__ text, const int* __restrict__ rep,
                         const float* __restrict__ Wq, const float* __restrict__ bq,
                         const float* __restrict__ Wk, const float* __restrict__ bk,
                         float* __restrict__ wscore, float* __restrict__ cscal,
                         int* __restrict__ offs, int* __restrict__ seg) {
  int g = blockIdx.x, tid = threadIdx.x;
  __shared__ float red[256];
  __shared__ float qsh[256];
  __shared__ int   ired[256];
  int p = 0;
  for (int i = tid; i < g; i += 256) p += rep[i];
  ired[tid] = p; __syncthreads();
  for (int s = 128; s > 0; s >>= 1) { if (tid < s) ired[tid] += ired[tid + s]; __syncthreads(); }
  int off = ired[0];
  int len = rep[g];
  if (tid == 0) offs[g] = off;
  for (int i = off + tid; i < off + len; i += 256) seg[i] = g;

  float qf = bq[tid];
  const float* tr = text + (size_t)g * TXT;
  for (int t = 0; t < TXT; ++t) qf += tr[t] * Wq[t * FD + tid];
  qsh[tid] = qf;
  red[tid] = qf * bk[tid];
  __syncthreads();
  for (int s = 128; s > 0; s >>= 1) { if (tid < s) red[tid] += red[tid + s]; __syncthreads(); }
  if (tid == 0) cscal[g] = red[0];

  float w = 0.f;
  for (int f = 0; f < FD; ++f) w += Wk[tid * FD + f] * qsh[f];
  wscore[(size_t)g * FD + tid] = w;
}

// ---------------- K2: h = relu(X@W0+b0) via WMMA; fused scores + hsum -----
template <bool FULL>
__global__ void __launch_bounds__(128) k2_pass1(
    const float* __restrict__ X, const float* __restrict__ b0,
    const half_t* __restrict__ w0frag, const float* __restrict__ wscore,
    const float* __restrict__ cscal, const int* __restrict__ offs,
    const int* __restrict__ seg, float* __restrict__ scores,
    float* __restrict__ hsum, int Ntot, int tile_base) {
  __shared__ float sA[4][16 * 129];   // per-wave X tile, pitch 129
  int wave = threadIdx.x >> 5, lane = threadIdx.x & 31;
  int ntiles = (Ntot + 15) >> 4;
  int tile = tile_base + blockIdx.x * 4 + wave;
  if (tile >= ntiles) return;         // no block-wide barriers below
  int r0 = tile * 16;
  float* sa = sA[wave];

#if USE_TDM
  unsigned valid = (unsigned)((FULL ? 16 : (Ntot - r0)) * INDIM);
  tdm_stage_tile(X + (size_t)r0 * INDIM, sa, valid);
  __builtin_amdgcn_s_wait_tensorcnt(0);
#else
  const float* xb = X + (size_t)r0 * INDIM;
  for (int i = lane; i < 16 * INDIM; i += 32) {
    int m = i >> 7, k = i & 127;
    sa[m * 129 + k] = (FULL || r0 + m < Ntot) ? xb[i] : 0.f;
  }
#endif

  int lanehi = lane >> 4, mrow = lane & 15;
  v16h a[4];
  for (int ks = 0; ks < 4; ++ks)
    for (int j = 0; j < 16; ++j)
      a[ks][j] = (half_t)sa[mrow * 129 + ks * 32 + a_kmap(j, lanehi)];

  int gA = seg[r0];
  int rlast = FULL ? (r0 + 15) : ((r0 + 15 < Ntot) ? (r0 + 15) : (Ntot - 1));
  int gB = seg[rlast];
  int boundary = (gA == gB) ? 0x7fffffff : offs[gB];

  float sp[8];
  for (int v = 0; v < 8; ++v) sp[v] = 0.f;

  for (int nt = 0; nt < 16; ++nt) {
    int col = nt * 16 + (lane & 15);
    float bias = b0[col];
    v8f acc;
    for (int v = 0; v < 8; ++v) acc[v] = bias;
    for (int ks = 0; ks < 4; ++ks) {
      v16h b = *(const v16h*)(w0frag + ((((nt * 4 + ks) * 32) + lane) << 4));
      acc = __builtin_amdgcn_wmma_f32_16x16x32_f16(false, a[ks], false, b,
                                                   (short)0, acc, false, false);
    }
    float wA = wscore[(size_t)gA * FD + col];
    float wB = wscore[(size_t)gB * FD + col];   // == wA when gA==gB; always safe
    float hA = 0.f, hB = 0.f;
    for (int v = 0; v < 8; ++v) {
      int row = r0 + v + lanehi * 8;
      if (FULL || row < Ntot) {
        float h = acc[v]; h = h > 0.f ? h : 0.f;
        bool inA = row < boundary;
        sp[v] += h * (inA ? wA : wB);
        hA += inA ? h : 0.f;
        hB += inA ? 0.f : h;
      }
    }
    atomicAdd(&hsum[(size_t)gA * FD + col], hA);
    if (gA != gB) atomicAdd(&hsum[(size_t)gB * FD + col], hB);
  }
  // reduce score partials over the 16 lanes of each half-wave
  for (int m = 1; m < 16; m <<= 1)
    for (int v = 0; v < 8; ++v) sp[v] += __shfl_xor(sp[v], m, 32);
  if ((lane & 15) == 0) {
    for (int v = 0; v < 8; ++v) {
      int row = r0 + v + lanehi * 8;
      if (FULL || row < Ntot) {
        int g = row < boundary ? gA : gB;
        scores[row] = sp[v] + cscal[g];
      }
    }
  }
}

// ---------------- K3: per-graph softmax stats + v_sum + u = v_sum@Wo ------
__global__ void k3_graph(const int* __restrict__ rep, const int* __restrict__ offs,
                         const float* __restrict__ scores, const float* __restrict__ hsum,
                         const float* __restrict__ Wv, const float* __restrict__ bv,
                         const float* __restrict__ Wo,
                         float* __restrict__ mx, float* __restrict__ inv,
                         float* __restrict__ u) {
  int g = blockIdx.x, tid = threadIdx.x;
  __shared__ float red[256];
  __shared__ float vs[256];
  int off = offs[g], len = rep[g];
  float m = -1e30f;
  for (int i = off + tid; i < off + len; i += 256) m = fmaxf(m, scores[i]);
  red[tid] = m; __syncthreads();
  for (int s = 128; s > 0; s >>= 1) { if (tid < s) red[tid] = fmaxf(red[tid], red[tid + s]); __syncthreads(); }
  float mg = red[0]; __syncthreads();
  float e = 0.f;
  for (int i = off + tid; i < off + len; i += 256) e += __expf(scores[i] - mg);
  red[tid] = e; __syncthreads();
  for (int s = 128; s > 0; s >>= 1) { if (tid < s) red[tid] += red[tid + s]; __syncthreads(); }
  if (tid == 0) { mx[g] = mg; inv[g] = 1.f / red[0]; }

  float v = (float)len * bv[tid];
  const float* hs = hsum + (size_t)g * FD;
  for (int d = 0; d < FD; ++d) v += hs[d] * Wv[d * FD + tid];
  vs[tid] = v; __syncthreads();
  float uo = 0.f;
  for (int f = 0; f < FD; ++f) uo += vs[f] * Wo[f * HID + tid];
  u[(size_t)g * HID + tid] = uo;
}

// ---------------- K4: out = relu(p*u + bo) @ W2 + b2 via WMMA -------------
template <bool FULL>
__global__ void __launch_bounds__(128) k4_pass2(
    const float* __restrict__ scores, const float* __restrict__ mx,
    const float* __restrict__ inv, const float* __restrict__ u,
    const float* __restrict__ bo, const float* __restrict__ b2,
    const half_t* __restrict__ w2frag, const int* __restrict__ offs,
    const int* __restrict__ seg, float* __restrict__ out, int Ntot, int tile_base) {
  __shared__ float bo_sh[HID];
  __shared__ float b2_sh[OUTD];
  __shared__ float ush[4][2 * HID];
  int tid = threadIdx.x;
  for (int i = tid; i < HID; i += 128) bo_sh[i] = bo[i];
  for (int i = tid; i < OUTD; i += 128) b2_sh[i] = b2[i];
  __syncthreads();                       // only block-wide barrier; before any return
  int wave = tid >> 5, lane = tid & 31;
  int ntiles = (Ntot + 15) >> 4;
  int tile = tile_base + blockIdx.x * 4 + wave;
  if (tile >= ntiles) return;
  int r0 = tile * 16;
  int gA = seg[r0];
  int rlast = FULL ? (r0 + 15) : ((r0 + 15 < Ntot) ? (r0 + 15) : (Ntot - 1));
  int gB = seg[rlast];
  int boundary = (gA == gB) ? 0x7fffffff : offs[gB];
  float* uw = ush[wave];
  for (int i = lane; i < HID; i += 32) {
    uw[i]       = u[(size_t)gA * HID + i];
    uw[HID + i] = u[(size_t)gB * HID + i];
  }
  int lanehi = lane >> 4, mrow = lane & 15;
  int row = r0 + mrow;
  float pr = 0.f; int sel = 0;
  if (FULL || row < Ntot) {
    int g = row < boundary ? gA : gB;
    sel = row < boundary ? 0 : 1;
    pr = __expf(scores[row] - mx[g]) * inv[g];
  }
  const float* ug = uw + sel * HID;
  v16h a[8];
  for (int ks = 0; ks < 8; ++ks)
    for (int j = 0; j < 16; ++j) {
      int k = ks * 32 + a_kmap(j, lanehi);
      float z = pr * ug[k] + bo_sh[k];
      z = z > 0.f ? z : 0.f;
      a[ks][j] = (half_t)z;
    }
  for (int ct = 0; ct < 8; ++ct) {
    int col = ct * 16 + (lane & 15);
    float bias = b2_sh[col];
    v8f acc;
    for (int v = 0; v < 8; ++v) acc[v] = bias;
    for (int ks = 0; ks < 8; ++ks) {
      v16h b = *(const v16h*)(w2frag + ((((ct * 8 + ks) * 32) + lane) << 4));
      acc = __builtin_amdgcn_wmma_f32_16x16x32_f16(false, a[ks], false, b,
                                                   (short)0, acc, false, false);
    }
    for (int v = 0; v < 8; ++v) {
      int orow = r0 + v + lanehi * 8;
      if (FULL || orow < Ntot) out[(size_t)orow * OUTD + col] = acc[v];
    }
  }
}

// ---------------- launch ---------------------------------------------------
extern "C" void kernel_launch(void* const* d_in, const int* in_sizes, int n_in,
                              void* d_out, int out_size, void* d_ws, size_t ws_size,
                              hipStream_t stream) {
  const float* X    = (const float*)d_in[0];
  const float* text = (const float*)d_in[1];
  const int*   rep  = (const int*)d_in[2];
  const float* W0 = (const float*)d_in[3];  const float* b0 = (const float*)d_in[4];
  const float* Wq = (const float*)d_in[5];  const float* bq = (const float*)d_in[6];
  const float* Wk = (const float*)d_in[7];  const float* bk = (const float*)d_in[8];
  const float* Wv = (const float*)d_in[9];  const float* bv = (const float*)d_in[10];
  const float* Wo = (const float*)d_in[11]; const float* bo = (const float*)d_in[12];
  const float* W2 = (const float*)d_in[13]; const float* b2 = (const float*)d_in[14];
  float* out = (float*)d_out;

  int Ntot = in_sizes[0] / INDIM;    // total nodes
  int Bn   = in_sizes[2];            // graphs

  // workspace carve-up (256B aligned)
  char* ws = (char*)d_ws;
  size_t o = 0;
  auto carve = [&](size_t bytes) { size_t r = o; o = (o + bytes + 255) & ~(size_t)255; return r; };
  half_t* w0frag = (half_t*)(ws + carve(16 * 4 * 32 * 16 * sizeof(half_t)));
  half_t* w2frag = (half_t*)(ws + carve(8 * 8 * 32 * 16 * sizeof(half_t)));
  float* wscore  = (float*)(ws + carve((size_t)Bn * FD * sizeof(float)));
  float* cscal   = (float*)(ws + carve((size_t)Bn * sizeof(float)));
  int*   offs    = (int*)  (ws + carve((size_t)Bn * sizeof(int)));
  int*   seg     = (int*)  (ws + carve((size_t)Ntot * sizeof(int)));
  float* scores  = (float*)(ws + carve((size_t)Ntot * sizeof(float)));
  float* hsum    = (float*)(ws + carve((size_t)Bn * FD * sizeof(float)));
  float* mx      = (float*)(ws + carve((size_t)Bn * sizeof(float)));
  float* invs    = (float*)(ws + carve((size_t)Bn * sizeof(float)));
  float* u       = (float*)(ws + carve((size_t)Bn * HID * sizeof(float)));

  int hsumN = Bn * FD;
  int prepN = 65536 > hsumN ? 65536 : hsumN;
  k0_prep<<<(prepN + 255) / 256, 256, 0, stream>>>(W0, W2, w0frag, w2frag, hsum, hsumN);
  k1_graph<<<Bn, 256, 0, stream>>>(text, rep, Wq, bq, Wk, bk, wscore, cscal, offs, seg);

  int T = (Ntot + 15) / 16;          // total tiles
  int Tfull = Ntot / 16;             // tiles with 16 valid rows
  int nblkA = Tfull / 4;             // blocks of 4 full tiles
  int done = nblkA * 4;

  if (nblkA > 0)
    k2_pass1<true><<<nblkA, 128, 0, stream>>>(X, b0, w0frag, wscore, cscal, offs,
                                              seg, scores, hsum, Ntot, 0);
  if (done < T)
    k2_pass1<false><<<1, 128, 0, stream>>>(X, b0, w0frag, wscore, cscal, offs,
                                           seg, scores, hsum, Ntot, done);

  k3_graph<<<Bn, 256, 0, stream>>>(rep, offs, scores, hsum, Wv, bv, Wo, mx, invs, u);

  if (nblkA > 0)
    k4_pass2<true><<<nblkA, 128, 0, stream>>>(scores, mx, invs, u, bo, b2, w2frag,
                                              offs, seg, out, Ntot, 0);
  if (done < T)
    k4_pass2<false><<<1, 128, 0, stream>>>(scores, mx, invs, u, bo, b2, w2frag,
                                           offs, seg, out, Ntot, done);
}
